// IntegratedFiberOpticsNN_9036611190934
// MI455X (gfx1250) — matrix operand order repaired and checked by reading, not compiled, not run on hardware
//
#include <hip/hip_runtime.h>
#include <hip/hip_bf16.h>
#include <math.h>

typedef __attribute__((ext_vector_type(16))) _Float16 v16h;
typedef __attribute__((ext_vector_type(8)))  float    v8f;

#define WG 256

// ---------------------------------------------------------------------------
// WMMA GEMM:  out[n,m] = sum_k A[m,k] * B(n,k)  (f16 in, f32 accum)
//   MODE 0: B(n,k) = Bt[n*Ks + k]                         (plain, Ks mult 32)
//   MODE 1: implicit im2col 3x3 pad1 from f16 NHWC        (Cin = pow2 >= 32)
//   MODE 2: implicit deconv(k4,s2,p1) phase taps          (Cin = pow2 >= 32)
// MT = 16-row accumulator tiles per wave (A padded to 64 rows, zero filled).
// All divides are shifts: pixShift=log2(H*W), wShift=log2(W), ciShift=log2(Cin).
// ---------------------------------------------------------------------------
struct GemmDesc {
  const _Float16* A;      // [ceil64(M), K] row-major, K mult of 32, zero padded
  const _Float16* Bsrc;   // MODE0: Bt[N,Ks]; MODE1/2: f16 NHWC [B,H,W,Cin]
  float*    outF;
  _Float16* outH;
  const float* bias;      // [M] or null
  const float* scale;     // [Bn*M] or null (per batch+channel)
  const float* colscale;  // [N] or null
  int M, N, K, Ks;
  int mode, Cin, H, W, phase;
  int pixShift, wShift, ciShift;
  int osF, ooF, osH, ooH;
  int relu, accum, ups;
};

template <int MODE, int MT>
__global__ __launch_bounds__(32) void k_gemm(GemmDesc d) {
  const int ntile = blockIdx.x;
  const int mbase0 = blockIdx.y * (16 * MT);
  const int lane = threadIdx.x & 31;
  const int hi = lane >> 4, lm = lane & 15;
  const int n = ntile * 16 + lm;
  const int nClamp = (n < d.N) ? n : (d.N - 1);
  int nb = 0, y = 0, x = 0;
  if (MODE != 0 || d.scale != nullptr || d.ups) {
    nb = nClamp >> d.pixShift;
    int rem = nClamp & ((1 << d.pixShift) - 1);
    y = rem >> d.wShift;
    x = rem & ((1 << d.wShift) - 1);
  }
  const _Float16* Ap0[MT];
#pragma unroll
  for (int t = 0; t < MT; t++)
    Ap0[t] = d.A + (size_t)(mbase0 + t * 16 + lm) * d.K + hi * 16;
  v8f acc[MT];
#pragma unroll
  for (int t = 0; t < MT; t++) acc[t] = (v8f){};

  if (MODE == 0) {
    const _Float16* bp = d.Bsrc + (size_t)nClamp * d.Ks + hi * 16;
    const _Float16* Ap[MT];
#pragma unroll
    for (int t = 0; t < MT; t++) Ap[t] = Ap0[t];
    for (int k0 = 0; k0 < d.K; k0 += 32) {
      v16h bb;
      ((uint4*)&bb)[0] = ((const uint4*)bp)[0];
      ((uint4*)&bb)[1] = ((const uint4*)bp)[1];
      bp += 32;
#pragma unroll
      for (int t = 0; t < MT; t++) {
        v16h a;
        ((uint4*)&a)[0] = *(const uint4*)(Ap[t]);
        ((uint4*)&a)[1] = *(const uint4*)(Ap[t] + 8);
        Ap[t] += 32;
        acc[t] = __builtin_amdgcn_wmma_f32_16x16x32_f16(false, a, false, bb,
                                                        (short)0, acc[t], false, false);
      }
    }
  } else {
    const int Cin = d.Cin, s = d.ciShift;
    const _Float16* cbase = d.Bsrc + (((size_t)(nb * d.H + y) * d.W + x) << s);
    const int NT = (MODE == 1) ? 9 : 4;
    const int py = d.phase >> 1, px = d.phase & 1;
    for (int tap = 0; tap < NT; tap++) {
      int dy, dx;
      if (MODE == 1) {
        dy = (tap >= 6) ? 1 : (tap >= 3) ? 0 : -1;
        dx = tap - (dy + 1) * 3 - 1;
      } else {
        int a_ = tap >> 1, b_ = tap & 1;
        dy = py ? (a_ ? 0 : 1) : (a_ ? -1 : 0);
        dx = px ? (b_ ? 0 : 1) : (b_ ? -1 : 0);
      }
      const int iy = y + dy, ix = x + dx;
      const bool valid = (iy >= 0) && (iy < d.H) && (ix >= 0) && (ix < d.W);
      if (!__any((int)valid)) continue;
      const _Float16* bp =
          (valid ? (cbase + (((ptrdiff_t)dy * d.W + dx) << s)) : d.Bsrc) + hi * 16;
      const _Float16* Ap[MT];
#pragma unroll
      for (int t = 0; t < MT; t++) Ap[t] = Ap0[t] + ((size_t)tap << s);
      if (__all((int)valid)) {
        for (int kk = 0; kk < Cin; kk += 32) {
          v16h bb;
          ((uint4*)&bb)[0] = ((const uint4*)bp)[0];
          ((uint4*)&bb)[1] = ((const uint4*)bp)[1];
          bp += 32;
#pragma unroll
          for (int t = 0; t < MT; t++) {
            v16h a;
            ((uint4*)&a)[0] = *(const uint4*)(Ap[t]);
            ((uint4*)&a)[1] = *(const uint4*)(Ap[t] + 8);
            Ap[t] += 32;
            acc[t] = __builtin_amdgcn_wmma_f32_16x16x32_f16(false, a, false, bb,
                                                            (short)0, acc[t], false, false);
          }
        }
      } else {
        for (int kk = 0; kk < Cin; kk += 32) {
          uint4 q0 = ((const uint4*)bp)[0];
          uint4 q1 = ((const uint4*)bp)[1];
          bp += 32;
          if (!valid) { q0.x = q0.y = q0.z = q0.w = 0u; q1.x = q1.y = q1.z = q1.w = 0u; }
          v16h bb;
          ((uint4*)&bb)[0] = q0;
          ((uint4*)&bb)[1] = q1;
#pragma unroll
          for (int t = 0; t < MT; t++) {
            v16h a;
            ((uint4*)&a)[0] = *(const uint4*)(Ap[t]);
            ((uint4*)&a)[1] = *(const uint4*)(Ap[t] + 8);
            Ap[t] += 32;
            acc[t] = __builtin_amdgcn_wmma_f32_16x16x32_f16(false, a, false, bb,
                                                            (short)0, acc[t], false, false);
          }
        }
      }
    }
  }

  if (n >= d.N) return;
  const float cs = d.colscale ? d.colscale[n] : 1.0f;
  size_t on;
  {
    int py = d.phase >> 1, px = d.phase & 1;
    if (d.ups)
      on = ((size_t)(nb * 2 * d.H + 2 * y + py)) * (size_t)(2 * d.W) + (size_t)(2 * x + px);
    else
      on = (size_t)n;
  }
  const size_t oF = on * (size_t)d.osF + d.ooF;
  const size_t oH = on * (size_t)d.osH + d.ooH;
#pragma unroll
  for (int t = 0; t < MT; t++) {
    int mb = mbase0 + t * 16 + hi * 8;
#pragma unroll
    for (int r = 0; r < 8; r++) {
      int m = mb + r;
      if (m >= d.M) break;
      float v = acc[t][r] * cs;
      if (d.bias) v += d.bias[m];
      if (d.scale) v *= d.scale[nb * d.M + m];
      if (d.accum) v += d.outF[oF + m];
      if (d.relu) v = fmaxf(v, 0.f);
      if (d.outF) d.outF[oF + m] = v;
      if (d.outH) d.outH[oH + m] = (_Float16)v;
    }
  }
}

// ---------------------------------------------------------------------------
__global__ void k_zero(float4* p, long long n4) {
  long long i = (long long)blockIdx.x * blockDim.x + threadIdx.x;
  long long st = (long long)gridDim.x * blockDim.x;
  float4 z; z.x = z.y = z.z = z.w = 0.f;
  for (; i < n4; i += st) p[i] = z;
}

__global__ void k_pack_convw(const float* w, _Float16* A, int M, int CiTot,
                             int ciOff, int Cl, int KH, int KW, int Kp) {
  long long i = (long long)blockIdx.x * blockDim.x + threadIdx.x;
  long long KL = (long long)KH * KW * Cl;
  if (i >= (long long)M * KL) return;
  int m = (int)(i / KL); int k = (int)(i - (long long)m * KL);
  int tap = k / Cl, ci = k - tap * Cl;
  A[(size_t)m * Kp + k] =
      (_Float16)w[(((size_t)m * CiTot + ciOff + ci) * KH + tap / KW) * KW + tap % KW];
}

__global__ void k_pack_deconvw(const float* w, _Float16* A, int Ci, int Co,
                               int phase, int Kp) {
  long long i = (long long)blockIdx.x * blockDim.x + threadIdx.x;
  int K = 4 * Ci;
  if (i >= (long long)Co * K) return;
  int m = (int)(i / K); int k = (int)(i - (long long)m * K);
  int tap = k / Ci, ci = k - tap * Ci;
  int a = tap >> 1, b_ = tap & 1, py = phase >> 1, px = phase & 1;
  int ky = py ? (a ? 2 : 0) : (a ? 3 : 1);
  int kx = px ? (b_ ? 2 : 0) : (b_ ? 3 : 1);
  A[(size_t)m * Kp + k] = (_Float16)w[(((size_t)ci * Co + m) * 4 + ky) * 4 + kx];
}

__global__ void k_pack_patterns(const float* pat, _Float16* A, int C, int Kp) {
  int p = blockIdx.x;
  __shared__ float red[WG];
  float s = 0;
  for (int c = threadIdx.x; c < C; c += blockDim.x) { float v = pat[(size_t)p * C + c]; s += v * v; }
  red[threadIdx.x] = s; __syncthreads();
  for (int st = WG / 2; st > 0; st >>= 1) {
    if (threadIdx.x < st) red[threadIdx.x] += red[threadIdx.x + st];
    __syncthreads();
  }
  float inv = 1.f / fmaxf(sqrtf(red[0]), 1e-8f);
  for (int c = threadIdx.x; c < C; c += blockDim.x)
    A[(size_t)p * Kp + c] = (_Float16)(pat[(size_t)p * C + c] * inv);
}

__global__ void k_pack_emb(const float* emb, _Float16* A) {
  int e = blockIdx.x;
  __shared__ float red[WG];
  float s = 0;
  for (int j = threadIdx.x; j < 8192; j += blockDim.x) { float v = emb[(size_t)e * 8192 + j]; s += v * v; }
  red[threadIdx.x] = s; __syncthreads();
  for (int st = WG / 2; st > 0; st >>= 1) {
    if (threadIdx.x < st) red[threadIdx.x] += red[threadIdx.x + st];
    __syncthreads();
  }
  float inv = 1.f / fmaxf(sqrtf(red[0]), 1e-8f);
  for (int j = threadIdx.x; j < 8192; j += blockDim.x) {
    int c = j & 127, sp = j >> 7;
    A[(size_t)e * 8192 + j] = (_Float16)(emb[(size_t)e * 8192 + c * 64 + sp] * inv);
  }
}

__global__ void k_im2col3(const float* x, _Float16* col, int Bn, int H, int W) {
  long long i = (long long)blockIdx.x * blockDim.x + threadIdx.x;
  long long tot = (long long)Bn * H * W * 27;
  if (i >= tot) return;
  int k = (int)(i % 27); long long n = i / 27;
  int xx = (int)(n % W); int yy = (int)((n / W) % H); int b = (int)(n / ((long long)H * W));
  int ci = k % 3, tap = k / 3;
  int sy = yy + tap / 3 - 1, sx = xx + tap % 3 - 1;
  float v = 0.f;
  if (sy >= 0 && sy < H && sx >= 0 && sx < W)
    v = x[(((size_t)b * 3 + ci) * H + sy) * W + sx];
  col[n * 32 + k] = (_Float16)v;
}

__global__ void k_stats(const float* Xp, int Cs, int H, int W,
                        long long sb, long long sc, long long sy, long long sx,
                        float* grads, float* poss) {
  int bc = blockIdx.x; int b = bc / Cs, c = bc % Cs;
  const float* base = Xp + (size_t)b * sb + (size_t)c * sc;
  float sg = 0, ss = 0, swy = 0, swx = 0;
  for (int p = threadIdx.x; p < H * W; p += blockDim.x) {
    int y = p / W, x = p - y * W;
    float v00 = 0, v01 = 0, v02 = 0, v10 = 0, v12 = 0, v20 = 0, v21 = 0, v22 = 0;
    bool yu = y > 0, yd = y < H - 1, xl = x > 0, xr = x < W - 1;
    if (yu && xl) v00 = base[(size_t)(y - 1) * sy + (size_t)(x - 1) * sx];
    if (yu)       v01 = base[(size_t)(y - 1) * sy + (size_t)x * sx];
    if (yu && xr) v02 = base[(size_t)(y - 1) * sy + (size_t)(x + 1) * sx];
    if (xl)       v10 = base[(size_t)y * sy + (size_t)(x - 1) * sx];
    if (xr)       v12 = base[(size_t)y * sy + (size_t)(x + 1) * sx];
    if (yd && xl) v20 = base[(size_t)(y + 1) * sy + (size_t)(x - 1) * sx];
    if (yd)       v21 = base[(size_t)(y + 1) * sy + (size_t)x * sx];
    if (yd && xr) v22 = base[(size_t)(y + 1) * sy + (size_t)(x + 1) * sx];
    float v11 = base[(size_t)y * sy + (size_t)x * sx];
    float gx = -v00 + v02 - 2.f * v10 + 2.f * v12 - v20 + v22;
    float gy = -v00 - 2.f * v01 - v02 + v20 + 2.f * v21 + v22;
    sg += sqrtf(gx * gx + gy * gy);
    ss += v11; swy += v11 * y; swx += v11 * x;
  }
  __shared__ float r[4][WG];
  r[0][threadIdx.x] = sg; r[1][threadIdx.x] = ss;
  r[2][threadIdx.x] = swy; r[3][threadIdx.x] = swx;
  __syncthreads();
  for (int st = WG / 2; st > 0; st >>= 1) {
    if (threadIdx.x < st)
      for (int j = 0; j < 4; j++) r[j][threadIdx.x] += r[j][threadIdx.x + st];
    __syncthreads();
  }
  if (threadIdx.x == 0) {
    float avg = r[0][0] / (float)(H * W);
    float tot = r[1][0] + 1e-8f;
    float ay = r[2][0] / tot, ax = r[3][0] / tot;
    grads[bc] = avg;
    poss[bc] = sqrtf((ay / H) * (ay / H) + (ax / W) * (ax / W));
  }
}

__global__ void k_finish_stats(const float* grads, const float* poss,
                               const float* gmod, const float* pmod,
                               const float* ginf, const float* pinf,
                               int Bn, int Cs, int Co, float* scale,
                               float* gml, float* pml) {
  __shared__ float rg[WG], rp[WG];
  __shared__ float gm_s, pm_s;
  for (int b = 0; b < Bn; b++) {
    float s = 0, p = 0;
    for (int c = threadIdx.x; c < Cs; c += blockDim.x) { s += grads[b * Cs + c]; p += poss[b * Cs + c]; }
    rg[threadIdx.x] = s; rp[threadIdx.x] = p; __syncthreads();
    for (int st = WG / 2; st > 0; st >>= 1) {
      if (threadIdx.x < st) { rg[threadIdx.x] += rg[threadIdx.x + st]; rp[threadIdx.x] += rp[threadIdx.x + st]; }
      __syncthreads();
    }
    if (threadIdx.x == 0) { gm_s = rg[0] / Cs; pm_s = rp[0] / Cs; gml[b] = gm_s; pml[b] = pm_s; }
    __syncthreads();
    float gf = gm_s * ginf[0], pf = pm_s * pinf[0];
    for (int c = threadIdx.x; c < Co; c += blockDim.x)
      scale[b * Co + c] = (1.f + gf * gmod[c]) * (1.f + pf * pmod[c]);
    __syncthreads();
  }
}

__global__ void k_invnorm(const float* F, int N, int C, float* out) {
  int n = blockIdx.x * blockDim.x + threadIdx.x;
  if (n >= N) return;
  const float* p = F + (size_t)n * C;
  float s = 0;
  for (int c = 0; c < C; c++) { float v = p[c]; s += v * v; }
  out[n] = 1.f / fmaxf(sqrtf(s), 1e-8f);
}

__global__ void k_pool16(const float* e1, int Bn, int H, int W, int C, _Float16* out) {
  long long i = (long long)blockIdx.x * blockDim.x + threadIdx.x;
  long long tot = (long long)Bn * 64 * C;
  if (i >= tot) return;
  int c = (int)(i % C); int r = (int)(i / C);
  int b = r / 64; int sp = r % 64; int oy = sp / 8, ox = sp % 8;
  float s = 0;
  for (int dy = 0; dy < 16; dy++)
    for (int dx = 0; dx < 16; dx++)
      s += e1[(((size_t)b * H + oy * 16 + dy) * W + ox * 16 + dx) * C + c];
  out[i] = (_Float16)(s / 256.f);
}

__global__ void k_flatnorm(const float* enc2f, float* inv) {
  int b = blockIdx.x;
  __shared__ float red[WG];
  float s = 0;
  for (int j = threadIdx.x; j < 8192; j += blockDim.x) { float v = enc2f[(size_t)b * 8192 + j]; s += v * v; }
  red[threadIdx.x] = s; __syncthreads();
  for (int st = WG / 2; st > 0; st >>= 1) {
    if (threadIdx.x < st) red[threadIdx.x] += red[threadIdx.x + st];
    __syncthreads();
  }
  if (threadIdx.x == 0) inv[b] = 1.f / fmaxf(sqrtf(red[0]), 1e-8f);
}

__global__ void k_argmax(const float* sims, int stride, int E,
                         float* oidx, float* osim, float* omeet) {
  int b = blockIdx.x;
  __shared__ float bv[WG]; __shared__ int bi[WG];
  float best = -1e30f; int bidx = 0;
  for (int e = threadIdx.x; e < E; e += blockDim.x) {
    float v = sims[(size_t)b * stride + e];
    if (v > best) { best = v; bidx = e; }
  }
  bv[threadIdx.x] = best; bi[threadIdx.x] = bidx; __syncthreads();
  for (int st = WG / 2; st > 0; st >>= 1) {
    if (threadIdx.x < st) {
      if (bv[threadIdx.x + st] > bv[threadIdx.x] ||
          (bv[threadIdx.x + st] == bv[threadIdx.x] && bi[threadIdx.x + st] < bi[threadIdx.x])) {
        bv[threadIdx.x] = bv[threadIdx.x + st]; bi[threadIdx.x] = bi[threadIdx.x + st];
      }
    }
    __syncthreads();
  }
  if (threadIdx.x == 0) {
    oidx[b] = (float)bi[0]; osim[b] = bv[0];
    omeet[b] = bv[0] > 0.7f ? 1.f : 0.f;
  }
}

__global__ void k_trend(const float* gml, const float* pml, const float* gt,
                        const float* pt, int Bn, float* trend) {
  int i = threadIdx.x;
  if (i >= Bn * 3) return;
  int b = i / 3, r = i % 3;
  float g = 0, p = 0;
  for (int l = 0; l < 4; l++) { g += gml[l * Bn + b]; p += pml[l * Bn + b]; }
  g *= 0.25f; p *= 0.25f;
  float z = gt[r * 2] * g + gt[r * 2 + 1] + pt[r * 2] * p + pt[r * 2 + 1];
  trend[i] = 1.f / (1.f + expf(-z));
}

__global__ void k_softmax_seg(const float* lg, int Bn, int H, int W, float* outSeg) {
  int n = blockIdx.x * blockDim.x + threadIdx.x;
  if (n >= Bn * H * W) return;
  float a = lg[(size_t)n * 3], b2 = lg[(size_t)n * 3 + 1], c = lg[(size_t)n * 3 + 2];
  float m = fmaxf(a, fmaxf(b2, c));
  float ea = expf(a - m), eb = expf(b2 - m), ec = expf(c - m);
  float s = ea + eb + ec;
  int b = n / (H * W), sp = n % (H * W);
  size_t o = (size_t)b * 3 * H * W + sp;
  outSeg[o] = ea / s; outSeg[o + (size_t)H * W] = eb / s; outSeg[o + 2 * (size_t)H * W] = ec / s;
}

__global__ void k_resize(const float* src, int Bn, int SH, int SW, int C,
                         int OH, int OW, float* out) {
  long long i = (long long)blockIdx.x * blockDim.x + threadIdx.x;
  long long tot = (long long)Bn * C * OH * OW;
  if (i >= tot) return;
  int ox = (int)(i % OW); long long t = i / OW;
  int oy = (int)(t % OH); t /= OH;
  int c = (int)(t % C); int b = (int)(t / C);
  float sy = (float)SH / OH, sx = (float)SW / OW;
  float cy = (oy + 0.5f) * sy - 0.5f, cx = (ox + 0.5f) * sx - 0.5f;
  int y0 = (int)ceilf(cy - sy), y1 = (int)floorf(cy + sy);
  int x0 = (int)ceilf(cx - sx), x1 = (int)floorf(cx + sx);
  if (y0 < 0) y0 = 0; if (y1 > SH - 1) y1 = SH - 1;
  if (x0 < 0) x0 = 0; if (x1 > SW - 1) x1 = SW - 1;
  float acc = 0, wsum = 0;
  for (int iy = y0; iy <= y1; iy++) {
    float wy = 1.f - fabsf(iy - cy) / sy;
    if (wy <= 0.f) continue;
    for (int ix = x0; ix <= x1; ix++) {
      float wx = 1.f - fabsf(ix - cx) / sx;
      if (wx <= 0.f) continue;
      float w = wy * wx;
      acc += w * src[(((size_t)b * SH + iy) * SW + ix) * C + c];
      wsum += w;
    }
  }
  out[i] = acc / fmaxf(wsum, 1e-12f);
}

__global__ void k_diff(const float* rec, const float* x, int Bn, int HW,
                       float* diff, float* mdiff) {
  int i = blockIdx.x * blockDim.x + threadIdx.x;
  if (i >= Bn * HW) return;
  int b = i / HW, sp = i % HW;
  float s = 0;
  for (int c = 0; c < 3; c++) {
    size_t o = ((size_t)b * 3 + c) * HW + sp;
    float d = fabsf(rec[o] - x[o]);
    diff[o] = d; s += d;
  }
  mdiff[i] = s * (1.f / 3.f);
}

__global__ void k_box_h(const float* A, const float* Bp, int mode, int Bn, int C,
                        int H, int W, float* out) {
  long long i = (long long)blockIdx.x * blockDim.x + threadIdx.x;
  long long tot = (long long)Bn * C * H * W;
  if (i >= tot) return;
  int x = (int)(i % W); long long t = i / W;
  int y = (int)(t % H); long long bc = t / H;
  float s = 0;
  for (int dx = -5; dx <= 5; dx++) {
    int xx = x + dx;
    if (xx < 0 || xx >= W) continue;
    size_t o = (bc * H + y) * (size_t)W + xx;
    float v = A[o];
    if (mode == 1) v *= v; else if (mode == 2) v *= Bp[o];
    s += v;
  }
  out[i] = s;
}

__global__ void k_box_v(const float* in, int Bn, int C, int H, int W, float* out) {
  long long i = (long long)blockIdx.x * blockDim.x + threadIdx.x;
  long long tot = (long long)Bn * C * H * W;
  if (i >= tot) return;
  int x = (int)(i % W); long long t = i / W;
  int y = (int)(t % H); long long bc = t / H;
  float s = 0;
  for (int dy = -5; dy <= 5; dy++) {
    int yy = y + dy;
    if (yy < 0 || yy >= H) continue;
    s += in[(bc * H + yy) * (size_t)W + x];
  }
  out[i] = s * (1.f / 121.f);
}

__global__ void k_anomaly(const float* mx, const float* my, const float* xx,
                          const float* yy, const float* xy, const float* mdiff,
                          const float* seg, const float* trend, int Bn, int H,
                          int W, float* outA) {
  int i = blockIdx.x * blockDim.x + threadIdx.x;
  if (i >= Bn * H * W) return;
  int b = i / (H * W), sp = i % (H * W);
  const float C1 = 0.0001f, C2 = 0.0009f;
  float ss = 0;
  for (int c = 0; c < 3; c++) {
    size_t o = ((size_t)b * 3 + c) * H * W + sp;
    float Mx = mx[o], My = my[o];
    float vxx = xx[o] - Mx * Mx, vyy = yy[o] - My * My, vxy = xy[o] - Mx * My;
    ss += ((2.f * Mx * My + C1) * (2.f * vxy + C2)) /
          ((Mx * Mx + My * My + C1) * (vxx + vyy + C2));
  }
  ss *= (1.f / 3.f);
  float gate = 0;
  for (int r = 0; r < 3; r++) {
    size_t o = ((size_t)b * 3 + r) * H * W + sp;
    gate += seg[o] * trend[b * 3 + r];
  }
  outA[i] = mdiff[i] * (1.f - ss) * gate;
}

// ---------------------------------------------------------------------------
static inline size_t alignUp(size_t v, size_t a) { return (v + a - 1) & ~(a - 1); }
static inline int ceil64i(int v) { return (v + 63) & ~63; }
static inline int ceil32i(int v) { return (v + 31) & ~31; }
static inline int ilog2i(int v) { int s = 0; while ((1 << s) < v) s++; return s; }

extern "C" void kernel_launch(void* const* d_in, const int* in_sizes, int n_in,
                              void* d_out, int out_size, void* d_ws, size_t ws_size,
                              hipStream_t stream) {
  (void)in_sizes; (void)n_in; (void)out_size; (void)ws_size;
  const int Bn = 2, H = 128, W = 128, HW = H * W, N = Bn * HW;
  const int chIn[4]  = {3, 64, 128, 256};
  const int chOut[4] = {64, 128, 256, 512};
  const int pats[4]  = {32, 48, 64, 96};

  const float* X = (const float*)d_in[0];
  const float *Lw[4], *Lb[4], *Lgm[4], *Lpm[4], *Lgi[4], *Lpi[4];
  for (int l = 0; l < 4; l++) {
    int b0 = 1 + 6 * l;
    Lw[l] = (const float*)d_in[b0];     Lb[l] = (const float*)d_in[b0 + 1];
    Lgm[l] = (const float*)d_in[b0 + 2]; Lpm[l] = (const float*)d_in[b0 + 3];
    Lgi[l] = (const float*)d_in[b0 + 4]; Lpi[l] = (const float*)d_in[b0 + 5];
  }
  const float *Cpat[4], *Cw1[4], *Cb1[4], *Cw2[4], *Cb2[4], *Cw3[4], *Cb3[4];
  for (int l = 0; l < 4; l++) {
    int b0 = 25 + 7 * l;
    Cpat[l] = (const float*)d_in[b0];     Cw1[l] = (const float*)d_in[b0 + 1];
    Cb1[l] = (const float*)d_in[b0 + 2];  Cw2[l] = (const float*)d_in[b0 + 3];
    Cb2[l] = (const float*)d_in[b0 + 4];  Cw3[l] = (const float*)d_in[b0 + 5];
    Cb3[l] = (const float*)d_in[b0 + 6];
  }
  const float* segw = (const float*)d_in[53];
  const float* segb = (const float*)d_in[54];
  const float* rw1 = (const float*)d_in[55]; const float* rb1 = (const float*)d_in[56];
  const float* rw2 = (const float*)d_in[57]; const float* rb2 = (const float*)d_in[58];
  const float* emb = (const float*)d_in[59];
  const float *Dw[4], *Db[4];
  for (int i = 0; i < 4; i++) { Dw[i] = (const float*)d_in[60 + 2 * i]; Db[i] = (const float*)d_in[61 + 2 * i]; }
  const float* gt = (const float*)d_in[68];
  const float* pt = (const float*)d_in[69];

  float* out = (float*)d_out;
  float* oSeg = out;
  float* oRec = out + 98304;
  float* oAnom = out + 196608;
  float* oDiff = out + 229376;
  float* oIdx = out + 327680; float* oSim = out + 327682; float* oMeet = out + 327684;

  char* ws = (char*)d_ws;
  size_t off = 0;
  auto alloc = [&](size_t bytes) -> void* {
    off = alignUp(off, 256);
    void* p = ws + off;
    off += bytes;
    return p;
  };

  _Float16* Aconv[4]; int KpConv[4];
  for (int l = 0; l < 4; l++) {
    KpConv[l] = (l == 0) ? 32 : 9 * chIn[l];
    Aconv[l] = (_Float16*)alloc((size_t)ceil64i(chOut[l]) * KpConv[l] * 2);
  }
  _Float16* col1 = (_Float16*)alloc((size_t)N * 32 * 2);
  float* fF[4]; _Float16* fH[4];
  for (int l = 0; l < 4; l++) {
    fF[l] = (float*)alloc((size_t)N * chOut[l] * 4);
    fH[l] = (_Float16*)alloc((size_t)N * chOut[l] * 2);
  }
  float* gradsB = (float*)alloc((size_t)Bn * 512 * 4);
  float* possB  = (float*)alloc((size_t)Bn * 512 * 4);
  float* scaleL[4];
  for (int l = 0; l < 4; l++) scaleL[l] = (float*)alloc((size_t)Bn * chOut[l] * 4);
  float* gml = (float*)alloc(4 * Bn * 4);
  float* pml = (float*)alloc(4 * Bn * 4);
  float* trend = (float*)alloc(Bn * 3 * 4);
  _Float16 *Apat[4], *Ac1a[4], *Ac1b[4], *Ac2[4], *Ac3[4];
  int Pp32[4];
  for (int l = 0; l < 4; l++) {
    int P = pats[l], Co = chOut[l];
    Pp32[l] = ceil32i(P);
    Apat[l] = (_Float16*)alloc((size_t)ceil64i(P) * Co * 2);
    Ac1a[l] = (_Float16*)alloc((size_t)128 * Co * 2);
    Ac1b[l] = (_Float16*)alloc((size_t)128 * Pp32[l] * 2);
    Ac2[l]  = (_Float16*)alloc((size_t)64 * 128 * 2);
    Ac3[l]  = (_Float16*)alloc((size_t)64 * 64 * 2);
  }
  _Float16* simsH = (_Float16*)alloc((size_t)N * 96 * 2);
  float* invn = (float*)alloc((size_t)N * 4);
  float* h1F = (float*)alloc((size_t)N * 128 * 4);
  _Float16* h1H = (_Float16*)alloc((size_t)N * 128 * 2);
  _Float16* h2H = (_Float16*)alloc((size_t)N * 64 * 2);
  _Float16* fusedH = (_Float16*)alloc((size_t)N * 32 * 2);
  float* seglog = (float*)alloc((size_t)N * 3 * 4);
  _Float16* Aseg = (_Float16*)alloc((size_t)64 * 32 * 2);
  _Float16* Aref1 = (_Float16*)alloc((size_t)256 * 4608 * 2);
  float* e1F = (float*)alloc((size_t)N * 256 * 4);
  _Float16* encH = (_Float16*)alloc((size_t)128 * 256 * 2);
  _Float16* Aref2 = (_Float16*)alloc((size_t)128 * 256 * 2);
  float* enc2F = (float*)alloc((size_t)128 * 128 * 4);
  _Float16* enc2H = (_Float16*)alloc((size_t)16 * 8192 * 2);
  float* invflat = (float*)alloc(16 * 4);
  _Float16* Aemb = (_Float16*)alloc((size_t)1024 * 8192 * 2);
  float* refsims = (float*)alloc((size_t)16 * 1024 * 4);
  const int dCi[4] = {512, 256, 128, 64}, dCo[4] = {256, 128, 64, 3};
  _Float16* Adec[4][4];
  for (int i = 0; i < 4; i++)
    for (int ph = 0; ph < 4; ph++)
      Adec[i][ph] = (_Float16*)alloc((size_t)ceil64i(dCo[i]) * 4 * dCi[i] * 2);
  _Float16* decH[3];
  decH[0] = (_Float16*)alloc((size_t)2 * 256 * 256 * 256 * 2);
  decH[1] = (_Float16*)alloc((size_t)2 * 512 * 512 * 128 * 2);
  decH[2] = (_Float16*)alloc((size_t)2 * 1024 * 1024 * 64 * 2);
  float* rec4 = (float*)alloc((size_t)2 * 2048 * 2048 * 3 * 4);
  float* mdiff = (float*)alloc((size_t)Bn * HW * 4);
  float* tmpB = (float*)alloc((size_t)Bn * 3 * HW * 4);
  float* mxF = (float*)alloc((size_t)Bn * 3 * HW * 4);
  float* myF = (float*)alloc((size_t)Bn * 3 * HW * 4);
  float* xxF = (float*)alloc((size_t)Bn * 3 * HW * 4);
  float* yyF = (float*)alloc((size_t)Bn * 3 * HW * 4);
  float* xyF = (float*)alloc((size_t)Bn * 3 * HW * 4);

  {
    long long n4 = (long long)((off + 15) / 16);
    long long b = (n4 + WG - 1) / WG;
    int blocks = (int)(b > 65536 ? 65536 : (b < 1 ? 1 : b));
    k_zero<<<dim3(blocks), dim3(WG), 0, stream>>>((float4*)ws, n4);
  }

  auto blk1d = [](long long tot) { return dim3((unsigned)((tot + WG - 1) / WG)); };
  auto rungemm = [&](const GemmDesc& d) {
    int MT = d.M >= 64 ? 4 : (d.M >= 32 ? 2 : 1);
    dim3 g((unsigned)((d.N + 15) / 16), (unsigned)((d.M + 16 * MT - 1) / (16 * MT)));
    dim3 b(32);
    int sel = d.mode * 3 + (MT == 4 ? 2 : MT == 2 ? 1 : 0);
    switch (sel) {
      case 0: k_gemm<0, 1><<<g, b, 0, stream>>>(d); break;
      case 1: k_gemm<0, 2><<<g, b, 0, stream>>>(d); break;
      case 2: k_gemm<0, 4><<<g, b, 0, stream>>>(d); break;
      case 3: k_gemm<1, 1><<<g, b, 0, stream>>>(d); break;
      case 4: k_gemm<1, 2><<<g, b, 0, stream>>>(d); break;
      case 5: k_gemm<1, 4><<<g, b, 0, stream>>>(d); break;
      case 6: k_gemm<2, 1><<<g, b, 0, stream>>>(d); break;
      case 7: k_gemm<2, 2><<<g, b, 0, stream>>>(d); break;
      default: k_gemm<2, 4><<<g, b, 0, stream>>>(d); break;
    }
  };

  // ---- pack all weight matrices ----
  for (int l = 0; l < 4; l++) {
    int Ci = chIn[l], Co = chOut[l], P = pats[l];
    k_pack_convw<<<blk1d((long long)Co * Ci * 9), WG, 0, stream>>>(
        Lw[l], Aconv[l], Co, Ci, 0, Ci, 3, 3, KpConv[l]);
    k_pack_patterns<<<dim3(P), WG, 0, stream>>>(Cpat[l], Apat[l], Co, Co);
    k_pack_convw<<<blk1d((long long)128 * Co), WG, 0, stream>>>(
        Cw1[l], Ac1a[l], 128, Co + P, 0, Co, 1, 1, Co);
    k_pack_convw<<<blk1d((long long)128 * P), WG, 0, stream>>>(
        Cw1[l], Ac1b[l], 128, Co + P, Co, P, 1, 1, Pp32[l]);
    k_pack_convw<<<blk1d((long long)64 * 128), WG, 0, stream>>>(
        Cw2[l], Ac2[l], 64, 128, 0, 128, 1, 1, 128);
    k_pack_convw<<<blk1d((long long)3 * 64), WG, 0, stream>>>(
        Cw3[l], Ac3[l], 3, 64, 0, 64, 1, 1, 64);
  }
  k_pack_convw<<<blk1d(3 * 12), WG, 0, stream>>>(segw, Aseg, 3, 12, 0, 12, 1, 1, 32);
  k_pack_convw<<<blk1d((long long)256 * 512 * 9), WG, 0, stream>>>(
      rw1, Aref1, 256, 512, 0, 512, 3, 3, 4608);
  k_pack_convw<<<blk1d((long long)128 * 256), WG, 0, stream>>>(
      rw2, Aref2, 128, 256, 0, 256, 1, 1, 256);
  k_pack_emb<<<dim3(1000), WG, 0, stream>>>(emb, Aemb);
  for (int i = 0; i < 4; i++)
    for (int ph = 0; ph < 4; ph++)
      k_pack_deconvw<<<blk1d((long long)dCo[i] * 4 * dCi[i]), WG, 0, stream>>>(
          Dw[i], Adec[i][ph], dCi[i], dCo[i], ph, 4 * dCi[i]);
  k_im2col3<<<blk1d((long long)N * 27), WG, 0, stream>>>(X, col1, Bn, H, W);

  // ---- backbone layers ----
  for (int l = 0; l < 4; l++) {
    int Ci = chIn[l], Co = chOut[l], P = pats[l];
    if (l == 0)
      k_stats<<<dim3(Bn * Ci), WG, 0, stream>>>(X, Ci, H, W,
          (long long)3 * HW, (long long)HW, (long long)W, 1LL, gradsB, possB);
    else
      k_stats<<<dim3(Bn * Ci), WG, 0, stream>>>(fF[l - 1], Ci, H, W,
          (long long)HW * Ci, 1LL, (long long)W * Ci, (long long)Ci, gradsB, possB);
    k_finish_stats<<<dim3(1), WG, 0, stream>>>(gradsB, possB, Lgm[l], Lpm[l],
        Lgi[l], Lpi[l], Bn, Ci, Co, scaleL[l], gml + l * Bn, pml + l * Bn);
    GemmDesc d{};
    d.A = Aconv[l]; d.bias = Lb[l]; d.scale = scaleL[l];
    d.M = Co; d.N = N; d.K = KpConv[l];
    d.H = H; d.W = W; d.pixShift = 14; d.wShift = 7;
    d.outF = fF[l]; d.outH = fH[l]; d.osF = Co; d.osH = Co;
    if (l == 0) { d.mode = 0; d.Bsrc = col1; d.Ks = 32; }
    else { d.mode = 1; d.Bsrc = fH[l - 1]; d.Cin = Ci; d.ciShift = ilog2i(Ci); }
    rungemm(d);
    k_invnorm<<<blk1d(N), WG, 0, stream>>>(fF[l], N, Co, invn);
    GemmDesc s{};
    s.A = Apat[l]; s.Bsrc = fH[l]; s.colscale = invn;
    s.M = P; s.N = N; s.K = Co; s.Ks = Co; s.mode = 0;
    s.outH = simsH; s.osH = Pp32[l];
    rungemm(s);
    GemmDesc p1{};
    p1.A = Ac1a[l]; p1.Bsrc = fH[l];
    p1.M = 128; p1.N = N; p1.K = Co; p1.Ks = Co; p1.mode = 0;
    p1.outF = h1F; p1.osF = 128;
    rungemm(p1);
    GemmDesc p2{};
    p2.A = Ac1b[l]; p2.Bsrc = simsH; p2.bias = Cb1[l];
    p2.M = 128; p2.N = N; p2.K = Pp32[l]; p2.Ks = Pp32[l]; p2.mode = 0;
    p2.outF = h1F; p2.osF = 128; p2.outH = h1H; p2.osH = 128;
    p2.accum = 1; p2.relu = 1;
    rungemm(p2);
    GemmDesc c2{};
    c2.A = Ac2[l]; c2.Bsrc = h1H; c2.bias = Cb2[l];
    c2.M = 64; c2.N = N; c2.K = 128; c2.Ks = 128; c2.mode = 0;
    c2.outH = h2H; c2.osH = 64; c2.relu = 1;
    rungemm(c2);
    GemmDesc c3{};
    c3.A = Ac3[l]; c3.Bsrc = h2H; c3.bias = Cb3[l];
    c3.M = 3; c3.N = N; c3.K = 64; c3.Ks = 64; c3.mode = 0;
    c3.outH = fusedH; c3.osH = 32; c3.ooH = 3 * l;
    rungemm(c3);
  }

  // ---- segmentation fusion + softmax ----
  {
    GemmDesc d{};
    d.A = Aseg; d.Bsrc = fusedH; d.bias = segb;
    d.M = 3; d.N = N; d.K = 32; d.Ks = 32; d.mode = 0;
    d.outF = seglog; d.osF = 3;
    rungemm(d);
    k_softmax_seg<<<blk1d(N), WG, 0, stream>>>(seglog, Bn, H, W, oSeg);
  }

  // ---- reference branch ----
  {
    GemmDesc d{};
    d.A = Aref1; d.Bsrc = fH[3]; d.bias = rb1;
    d.M = 256; d.N = N; d.K = 4608; d.mode = 1; d.Cin = 512; d.ciShift = 9;
    d.H = H; d.W = W; d.pixShift = 14; d.wShift = 7;
    d.outF = e1F; d.osF = 256; d.relu = 1;
    rungemm(d);
    k_pool16<<<blk1d((long long)Bn * 64 * 256), WG, 0, stream>>>(e1F, Bn, H, W, 256, encH);
    GemmDesc e{};
    e.A = Aref2; e.Bsrc = encH; e.bias = rb2;
    e.M = 128; e.N = Bn * 64; e.K = 256; e.Ks = 256; e.mode = 0;
    e.outF = enc2F; e.osF = 128; e.outH = enc2H; e.osH = 128; e.relu = 1;
    rungemm(e);
    k_flatnorm<<<dim3(Bn), WG, 0, stream>>>(enc2F, invflat);
    GemmDesc g{};
    g.A = Aemb; g.Bsrc = enc2H; g.colscale = invflat;
    g.M = 1000; g.N = Bn; g.K = 8192; g.Ks = 8192; g.mode = 0;
    g.outF = refsims; g.osF = 1024;
    rungemm(g);
    k_argmax<<<dim3(Bn), WG, 0, stream>>>(refsims, 1024, 1000, oIdx, oSim, oMeet);
  }

  // ---- decoder ----
  {
    const _Float16* din = fH[3];
    int Hi = H, Wi = W;
    for (int i = 0; i < 4; i++) {
      for (int ph = 0; ph < 4; ph++) {
        GemmDesc d{};
        d.A = Adec[i][ph]; d.Bsrc = din; d.bias = Db[i];
        d.M = dCo[i]; d.N = Bn * Hi * Wi; d.K = 4 * dCi[i];
        d.mode = 2; d.Cin = dCi[i]; d.ciShift = ilog2i(dCi[i]);
        d.H = Hi; d.W = Wi; d.pixShift = ilog2i(Hi * Wi); d.wShift = ilog2i(Wi);
        d.phase = ph; d.ups = 1; d.relu = (i < 3);
        if (i < 3) { d.outH = decH[i]; d.osH = dCo[i]; }
        else { d.outF = rec4; d.osF = 3; }
        rungemm(d);
      }
      if (i < 3) din = decH[i];
      Hi *= 2; Wi *= 2;
    }
    k_resize<<<blk1d((long long)Bn * 3 * HW), WG, 0, stream>>>(
        rec4, Bn, 2048, 2048, 3, H, W, oRec);
  }

  // ---- diff, SSIM, trend-gated anomaly ----
  k_diff<<<blk1d((long long)Bn * HW), WG, 0, stream>>>(oRec, X, Bn, HW, oDiff, mdiff);
  k_trend<<<dim3(1), dim3(32), 0, stream>>>(gml, pml, gt, pt, Bn, trend);
  long long nc = (long long)Bn * 3 * HW;
  k_box_h<<<blk1d(nc), WG, 0, stream>>>(oRec, nullptr, 0, Bn, 3, H, W, tmpB);
  k_box_v<<<blk1d(nc), WG, 0, stream>>>(tmpB, Bn, 3, H, W, mxF);
  k_box_h<<<blk1d(nc), WG, 0, stream>>>(X, nullptr, 0, Bn, 3, H, W, tmpB);
  k_box_v<<<blk1d(nc), WG, 0, stream>>>(tmpB, Bn, 3, H, W, myF);
  k_box_h<<<blk1d(nc), WG, 0, stream>>>(oRec, nullptr, 1, Bn, 3, H, W, tmpB);
  k_box_v<<<blk1d(nc), WG, 0, stream>>>(tmpB, Bn, 3, H, W, xxF);
  k_box_h<<<blk1d(nc), WG, 0, stream>>>(X, nullptr, 1, Bn, 3, H, W, tmpB);
  k_box_v<<<blk1d(nc), WG, 0, stream>>>(tmpB, Bn, 3, H, W, yyF);
  k_box_h<<<blk1d(nc), WG, 0, stream>>>(oRec, X, 2, Bn, 3, H, W, tmpB);
  k_box_v<<<blk1d(nc), WG, 0, stream>>>(tmpB, Bn, 3, H, W, xyF);
  k_anomaly<<<blk1d((long long)Bn * HW), WG, 0, stream>>>(
      mxF, myF, xxF, yyF, xyF, mdiff, oSeg, trend, Bn, H, W, oAnom);
}
